// EfmLSTM_214748365199
// MI455X (gfx1250) — matrix operand order repaired and checked by reading, not compiled, not run on hardware
//
#include <hip/hip_runtime.h>
#include <hip/hip_bf16.h>

// ---------------------------------------------------------------------------
// EfmLSTM on MI455X (gfx1250, wave32, WMMA f32_16x16x32_f16)
//
// B=64, T=1024, F=256, U=512, 3U=1536 gate columns (i,c,o), SIG_DIM=31.
//
//   k0 pack     : recurrent_kernel & input_kernel f32 -> f16 B-fragments
//   k1 f+init   : f_pre (signatures@forget_kernel+b_f) -> f16 D-fragments,
//                 zero h-buffer0 and barrier counters
//   k2 xproj    : x_all = inputs@input_kernel via WMMA, stored as f16
//                 accumulator-order fragments
//   k3 recur    : 16 persistent WGs, W-slice (96KB f16) in LDS, 1024 steps,
//                 per-step async Global->LDS h staging (ASYNCcnt) +
//                 software release/acquire barrier
// ---------------------------------------------------------------------------

typedef _Float16 v16h __attribute__((ext_vector_type(16)));
typedef _Float16 v8h  __attribute__((ext_vector_type(8)));
typedef float    v8f  __attribute__((ext_vector_type(8)));
typedef int      v4i  __attribute__((vector_size(16)));

#define _AS1 __attribute__((address_space(1)))
#define _AS3 __attribute__((address_space(3)))

#define B_   64
#define T_   1024
#define F_   256
#define U_   512
#define G3_  (3*U_)          // 1536
#define NW   16              // recurrence workgroups (unit slices of 32)
#define KC_R (U_/32)         // 16 k-chunks of 32 over h-dim
#define KC_X (F_/32)         // 8  k-chunks of 32 over feature dim

// workspace section sizes
#define WPACK_CNT   (NW*KC_R*6*32)                 // v16h count = 49152
#define XKPACK_CNT  (NW*KC_X*6*32)                 // v16h count = 24576
#define HBUF_CNT    (KC_R*4*32)                    // v16h per buffer = 2048
#define WPACK_BYTES  ((size_t)WPACK_CNT*32)        // 1.5 MB
#define XKPACK_BYTES ((size_t)XKPACK_CNT*32)       // 768 KB
#define HBUF_BYTES   ((size_t)2*HBUF_CNT*32)       // 128 KB (double buffered)
#define BAR_BYTES    ((size_t)T_*4)                // 4 KB
#define FPACK_CNT    ((size_t)T_*NW*4*2*32)        // v8h count = 4,194,304
#define XALL_CNT     ((size_t)T_*NW*4*6*32)        // v8h count = 12,582,912
#define FPACK_BYTES  (FPACK_CNT*16)                // 64 MB
#define XALL_BYTES   (XALL_CNT*16)                 // 192 MB

#if __has_builtin(__builtin_amdgcn_global_load_async_to_lds_b128)
#define HAVE_ASYNC_LDS 1
#else
#define HAVE_ASYNC_LDS 0
#endif

__device__ __forceinline__ float sigmoidf(float x) {
  return 1.0f / (1.0f + __expf(-x));
}

__device__ __forceinline__ v8f wmma_f16(v16h a, v16h b, v8f c) {
  // D = A(16x32 f16) x B(32x16 f16) + C(16x16 f32)
  return __builtin_amdgcn_wmma_f32_16x16x32_f16(
      /*neg_a=*/false, a, /*neg_b=*/false, b,
      /*c_mod=*/(short)0, c, /*reuse_a=*/false, /*reuse_b=*/false);
}

// A-fragment element->K map: lane<16 holds K={0..7,16..23}, lane>=16 {8..15,24..31}
__device__ __forceinline__ int a_kmap(int q, int lane) {
  return ((q >> 3) << 4) + ((lane >> 4) << 3) + (q & 7);
}
// B-fragment element->K map: lane<16 holds K={0..7,16..23}, lane>=16 {8..15,24..31}
__device__ __forceinline__ int b_kmap(int q, int lane) {
  return ((q >> 3) << 4) + ((lane >> 4) << 3) + (q & 7);
}

// ---------------------------------------------------------------------------
// k0: pack recurrent_kernel [512,1536] and input_kernel [256,1536] into
// f16 B-fragments. Column group for workgroup w, local tile j (0..5):
//   gate g=j/2 (i,c,o), unit-half = j&1, global col = 512*g + 32*w + 16*(j&1) + n
// wpack layout : [w][kc(16)][j(6)][lane(32)] x v16h
// xkpack layout: [w][kc(8) ][j(6)][lane(32)] x v16h
// ---------------------------------------------------------------------------
__global__ void pack_kernel(const float* __restrict__ Rk,
                            const float* __restrict__ Xk,
                            v16h* __restrict__ wpack,
                            v16h* __restrict__ xkpack) {
  int id = blockIdx.x * 256 + threadIdx.x;
  if (id < WPACK_CNT) {
    int lane = id & 31; int tmp = id >> 5;
    int j = tmp % 6; tmp /= 6;
    int kc = tmp % KC_R; int w = tmp / KC_R;
    int col = (j >> 1) * U_ + 32 * w + ((j & 1) << 4) + (lane & 15);
    v16h frag;
    #pragma unroll
    for (int q = 0; q < 16; q++) {
      int k = kc * 32 + b_kmap(q, lane);
      frag[q] = (_Float16)Rk[(long long)k * G3_ + col];
    }
    wpack[id] = frag;
  } else if (id < WPACK_CNT + XKPACK_CNT) {
    int id2 = id - WPACK_CNT;
    int lane = id2 & 31; int tmp = id2 >> 5;
    int j = tmp % 6; tmp /= 6;
    int kc = tmp % KC_X; int w = tmp / KC_X;
    int col = (j >> 1) * U_ + 32 * w + ((j & 1) << 4) + (lane & 15);
    v16h frag;
    #pragma unroll
    for (int q = 0; q < 16; q++) {
      int k = kc * 32 + b_kmap(q, lane);
      frag[q] = (_Float16)Xk[(long long)k * G3_ + col];
    }
    xkpack[id2] = frag;
  }
}

// ---------------------------------------------------------------------------
// k1: f_pre in C/D-fragment order + zero-init of h-buffer0 / barrier counters.
// fpack layout: [t][w][mtile(4)][uhalf(2)][lane(32)] x v8h ; element r maps to
//   row b = mtile*16 + 8*(lane>=16) + r, unit u = 32*w + 16*uhalf + lane%16.
// ---------------------------------------------------------------------------
__global__ void fproj_init_kernel(const float* __restrict__ sig,
                                  const float* __restrict__ FK,
                                  const float* __restrict__ bias,
                                  v8h* __restrict__ fpack,
                                  unsigned int* __restrict__ hzero, // 16384 dw
                                  unsigned int* __restrict__ bar) { // 1024 dw
  int id = blockIdx.x * 256 + threadIdx.x; // 4,194,304 total
  if (id < 16384) hzero[id] = 0u;
  if (id < T_)    bar[id]   = 0u;

  int lane = id & 31; int tmp = id >> 5;
  int uh = tmp & 1;  tmp >>= 1;
  int mt = tmp & 3;  tmp >>= 2;
  int w  = tmp & 15; int t = tmp >> 4;

  int u = 32 * w + (uh << 4) + (lane & 15);
  float bf = bias[U_ + u];
  float fk[31];
  #pragma unroll
  for (int d = 0; d < 31; d++) fk[d] = FK[d * U_ + u];

  int rbase = mt * 16 + ((lane >> 4) << 3);
  v8h res;
  #pragma unroll
  for (int r = 0; r < 8; r++) {
    const float* sp = sig + ((long long)(rbase + r) * T_ + t) * 31;
    float s = bf;
    #pragma unroll
    for (int d = 0; d < 31; d++) s += sp[d] * fk[d];
    res[r] = (_Float16)s;
  }
  fpack[id] = res;
}

// ---------------------------------------------------------------------------
// k2: x_all = inputs @ input_kernel via WMMA. One block per timestep t.
// A (64x256 for this t) staged in LDS as f16 A-fragments; B from global xkpack.
// Output stored directly in accumulator order:
//   xall layout: [t][w][mtile(4)][j(6)][lane(32)] x v8h
// ---------------------------------------------------------------------------
__global__ void __launch_bounds__(256) xproj_kernel(
    const float* __restrict__ X, const v16h* __restrict__ xkpack,
    v8h* __restrict__ xall) {
  __shared__ v16h alds[4 * KC_X * 32]; // 32 KB
  int t = blockIdx.x;
  int tid = threadIdx.x;

  #pragma unroll
  for (int i = 0; i < 4; i++) {
    int idx = tid + 256 * i;
    int lane = idx & 31; int tmp = idx >> 5;
    int kc = tmp % KC_X; int mt = tmp / KC_X;
    int b = mt * 16 + (lane & 15);
    const float* xp = X + ((long long)b * T_ + t) * F_;
    v16h frag;
    #pragma unroll
    for (int q = 0; q < 16; q++) frag[q] = (_Float16)xp[kc * 32 + a_kmap(q, lane)];
    alds[(mt * KC_X + kc) * 32 + lane] = frag;
  }
  __syncthreads();

  int wave = tid >> 5, lane = tid & 31;
  for (int s = 0; s < 48; s++) {
    int tau = wave * 48 + s;                 // 0..383 over (w, mt, j)
    int w = tau / 24; int rem = tau % 24;
    int mt = rem / 6; int j = rem % 6;
    v8f acc = {};
    #pragma unroll
    for (int kc = 0; kc < KC_X; kc++) {
      v16h a = alds[(mt * KC_X + kc) * 32 + lane];
      v16h b = xkpack[((w * KC_X + kc) * 6 + j) * 32 + lane];
      acc = wmma_f16(a, b, acc);
    }
    v8h o;
    #pragma unroll
    for (int e = 0; e < 8; e++) o[e] = (_Float16)acc[e];
    xall[((((long long)t * NW + w) * 4 + mt) * 6 + j) * 32 + lane] = o;
  }
}

// ---------------------------------------------------------------------------
// k3: persistent recurrence. 16 blocks x 256 threads. Block w owns units
// [32w,32w+32): its 96 gate columns (W slice, 96 KB f16) live in LDS.
// Per step: async-stage the full previous-h fragment buffer (64 KB) into LDS
// (GLOBAL_LOAD_ASYNC_TO_LDS_B128, ASYNCcnt), then a fully unrolled K=512
// WMMA chain (48 v_wmma per wave) reading A and B from LDS. Cell state in
// registers; fresh h transposed through 4 KB LDS into the other global
// buffer; release/acquire software barrier across the 16 WGs per step.
// ---------------------------------------------------------------------------
__global__ void __launch_bounds__(256, 1) recur_kernel(
    const v16h* __restrict__ wpack, const v8h* __restrict__ xall,
    const v8h* __restrict__ fpack, const float* __restrict__ bias,
    v16h* __restrict__ hbuf, unsigned int* __restrict__ bar,
    float* __restrict__ out) {
  __shared__ v16h w_lds[KC_R * 6 * 32];   // 96 KB  B-fragments
  __shared__ v16h h_in[KC_R * 4 * 32];    // 64 KB  staged A-fragments
  __shared__ _Float16 h_lds[64 * 32];     //  4 KB  h transpose scratch

  int w = blockIdx.x, tid = threadIdx.x;
  {
    const v16h* src = wpack + (long long)w * (KC_R * 6 * 32);
    for (int i = tid; i < KC_R * 6 * 32; i += 256) w_lds[i] = src[i];
  }
  int wave = tid >> 5, lane = tid & 31;
  int mtile = wave & 3, uh = wave >> 2;
  int u = 32 * w + (uh << 4) + (lane & 15);
  float bi = bias[u];
  float bc = bias[2 * U_ + u];
  float bo = bias[3 * U_ + u];
  float cst[8];
  #pragma unroll
  for (int r = 0; r < 8; r++) cst[r] = 0.0f;
  int rbase = mtile * 16 + ((lane >> 4) << 3);
  __syncthreads();

  for (int t = 0; t < T_; t++) {
    // ---- stage previous h (A-fragments) into LDS --------------------------
    const v16h* hin = hbuf + (t & 1) * HBUF_CNT;
#if HAVE_ASYNC_LDS
    {
      _AS1 v4i* g = (_AS1 v4i*)const_cast<v16h*>(hin);
      _AS3 v4i* l = (_AS3 v4i*)h_in;
      for (int i = tid; i < 4096; i += 256) {  // 4096 x 16B = 64 KB
        __builtin_amdgcn_global_load_async_to_lds_b128(g + i, l + i, 0, 0);
      }
#if __has_builtin(__builtin_amdgcn_s_wait_asynccnt)
      __builtin_amdgcn_s_wait_asynccnt(0);
#else
      asm volatile("s_wait_asynccnt 0x0" ::: "memory");
#endif
    }
#else
    for (int i = tid; i < KC_R * 4 * 32; i += 256) h_in[i] = hin[i];
#endif
    __syncthreads();

    // ---- gate bases for this step (overlap with WMMA), prefetch next ------
    long long xb = ((((long long)t * NW + w) * 4 + mtile) * 6) * 32 + lane;
    v8h xi = xall[xb + (long long)(uh    ) * 32];
    v8h xc = xall[xb + (long long)(2 + uh) * 32];
    v8h xo = xall[xb + (long long)(4 + uh) * 32];
    v8h fp = fpack[((((long long)t * NW + w) * 4 + mtile) * 2 + uh) * 32 + lane];
    if (t + 1 < T_) {
      long long xb2 = xb + (long long)NW * 4 * 6 * 32;
      __builtin_prefetch(&xall[xb2], 0, 1);
      __builtin_prefetch(&xall[xb2 + (long long)(2 + uh) * 32], 0, 1);
      __builtin_prefetch(&xall[xb2 + (long long)(4 + uh) * 32], 0, 1);
    }

    // ---- K=512 WMMA chain: 16 k-chunks x (i,c,o) --------------------------
    v8f ai = {}, ac = {}, ao = {};
    #pragma unroll
    for (int kc = 0; kc < KC_R; kc++) {
      v16h a = h_in[(kc * 4 + mtile) * 32 + lane];
      ai = wmma_f16(a, w_lds[(kc * 6 + uh    ) * 32 + lane], ai);
      ac = wmma_f16(a, w_lds[(kc * 6 + 2 + uh) * 32 + lane], ac);
      ao = wmma_f16(a, w_lds[(kc * 6 + 4 + uh) * 32 + lane], ao);
    }

    // ---- gates, cell update, h ------------------------------------------
    #pragma unroll
    for (int r = 0; r < 8; r++) {
      float iv = sigmoidf(ai[r] + (float)xi[r] + bi);
      float cv = tanhf  (ac[r] + (float)xc[r] + bc);
      float ov = sigmoidf(ao[r] + (float)xo[r] + bo);
      float fv = sigmoidf((float)fp[r]);
      cst[r] = fv * cst[r] + iv * cv;
      float hv = ov * tanhf(cst[r]);
      h_lds[(rbase + r) * 32 + (uh << 4) + (lane & 15)] = (_Float16)hv;
      if (t == T_ - 1) out[(rbase + r) * U_ + u] = hv;
    }
    __syncthreads();

    // ---- transpose h slice -> A-fragments for k-chunk w, all 4 mtiles -----
    v16h* hout = hbuf + ((t + 1) & 1) * HBUF_CNT;
    if (tid < 128) {
      int pl = tid & 31, pm = tid >> 5;
      int row = pm * 16 + (pl & 15);
      v16h hv16;
      #pragma unroll
      for (int q = 0; q < 16; q++) hv16[q] = h_lds[row * 32 + a_kmap(q, pl)];
      hout[(w * 4 + pm) * 32 + pl] = hv16;
    }
    __threadfence();
    __syncthreads();
    if (tid == 0) {
      __hip_atomic_fetch_add(&bar[t], 1u, __ATOMIC_ACQ_REL, __HIP_MEMORY_SCOPE_AGENT);
      while (__hip_atomic_load(&bar[t], __ATOMIC_ACQUIRE, __HIP_MEMORY_SCOPE_AGENT) < NW) {
        __builtin_amdgcn_s_sleep(1);
      }
    }
    __syncthreads();
    __threadfence();
  }
}

// ---------------------------------------------------------------------------
extern "C" void kernel_launch(void* const* d_in, const int* in_sizes, int n_in,
                              void* d_out, int out_size, void* d_ws, size_t ws_size,
                              hipStream_t stream) {
  const float* inputs = (const float*)d_in[0]; // [64,1024,256]
  const float* sig    = (const float*)d_in[1]; // [64,1024,31]
  const float* fk     = (const float*)d_in[2]; // [31,512]
  const float* xk     = (const float*)d_in[3]; // [256,1536]
  const float* rk     = (const float*)d_in[4]; // [512,1536]
  const float* bias   = (const float*)d_in[5]; // [2048]
  (void)in_sizes; (void)n_in; (void)out_size; (void)ws_size;

  char* p = (char*)d_ws;
  v16h* wpack  = (v16h*)p; p += WPACK_BYTES;
  v16h* xkpack = (v16h*)p; p += XKPACK_BYTES;
  v16h* hbuf   = (v16h*)p; p += HBUF_BYTES;
  unsigned int* bar = (unsigned int*)p; p += BAR_BYTES;
  v8h* fpack = (v8h*)p; p += FPACK_BYTES;
  v8h* xall  = (v8h*)p; p += XALL_BYTES;

  pack_kernel<<<(WPACK_CNT + XKPACK_CNT) / 256, 256, 0, stream>>>(rk, xk, wpack, xkpack);
  fproj_init_kernel<<<(int)(FPACK_CNT / 256), 256, 0, stream>>>(
      sig, fk, bias, fpack, (unsigned int*)hbuf, bar);
  xproj_kernel<<<T_, 256, 0, stream>>>(inputs, xkpack, xall);
  recur_kernel<<<NW, 256, 0, stream>>>(wpack, xall, fpack, bias, hbuf, bar,
                                       (float*)d_out);
}
